// GNN_8555574853744
// MI455X (gfx1250) — matrix operand order repaired and checked by reading, not compiled, not run on hardware
//
#include <hip/hip_runtime.h>
#include <hip/hip_bf16.h>

typedef __attribute__((ext_vector_type(16))) _Float16 v16h;
typedef __attribute__((ext_vector_type(8)))  _Float16 v8h;
typedef __attribute__((ext_vector_type(8)))  float    v8f;

#define NEG_SLOPE 0.2f

// ---------- float atomic-max via monotone uint encoding ----------
__device__ __forceinline__ unsigned fenc(float f) {
    unsigned u = __float_as_uint(f);
    return (u & 0x80000000u) ? ~u : (u | 0x80000000u);
}
__device__ __forceinline__ float fdec(unsigned u) {
    return (u & 0x80000000u) ? __uint_as_float(u ^ 0x80000000u)
                             : __uint_as_float(~u);
}
#define ENC_NEGINF 0x007FFFFFu   // fenc(-inf)

// ---------- per-edge alpha (recomputed in each pass; 8-FMA dot) ----------
__device__ __forceinline__ float edge_alpha(int e, int E,
        const int* __restrict__ src, const int* __restrict__ dst,
        const float* __restrict__ ea, const float* __restrict__ cvec,
        float aeloop, const float* __restrict__ as_, const float* __restrict__ ad_,
        int* ps, int* pd) {
    int s, d; float ae;
    if (e < E) {
        s = src[e]; d = dst[e];
        const float* ep = ea + (size_t)e * 8;
        ae = 0.f;
#pragma unroll
        for (int k = 0; k < 8; ++k) ae += ep[k] * cvec[k];
    } else {
        s = d = e - E;
        ae = aeloop;
    }
    *ps = s; *pd = d;
    float a = as_[s] + ad_[d] + ae;
    return (a > 0.f) ? a : NEG_SLOPE * a;
}

// ---------- pack W (f32, K x Nc row-major) into f16 fragment layout --------
// Wp[(k>>5)][col][k&31]  -> B fragment for a wave is 2 contiguous b128 loads.
__global__ void pack_w_f16(const float* __restrict__ W, _Float16* __restrict__ Wp,
                           int K, int Nc) {
    int i = blockIdx.x * blockDim.x + threadIdx.x;
    if (i >= K * Nc) return;
    int k = i / Nc, col = i - k * Nc;
    Wp[((size_t)(k >> 5) * Nc + col) * 32 + (k & 31)] = (_Float16)W[i];
}

// ---------- GEMM: C[M,Nc] = A[M,K] * Wp(K,Nc packed f16), f16 WMMA core ----
// block = 128 threads (4 waves), one 16-row M tile per block.
// Each wave: 16x64 output (4 WMMA accumulators). A tile staged via LDS as f16.
// B fragments prefetched in one clause, then 4 back-to-back WMMAs.
__global__ void gemm_wmma_f16(const float* __restrict__ A, const _Float16* __restrict__ Wp,
                              float* __restrict__ C, int M, int K, int Nc) {
    __shared__ _Float16 As[16][34];             // 16x32 tile, padded (bank-stagger)
    const int tid  = threadIdx.x;
    const int wave = tid >> 5;
    const int lane = tid & 31;
    const int lg   = lane >> 4;                 // lane group 0/1
    const int lr   = lane & 15;
    const int tileM = blockIdx.x * 16;
    const int tileN = wave * 64;
    const bool active = (tileN < Nc);

    v8f acc[4] = {};

    for (int k0 = 0; k0 < K; k0 += 32) {
        // cooperative, coalesced A-tile load (f32 -> f16 in LDS)
        for (int idx = tid; idx < 16 * 32; idx += 128) {
            int r  = idx >> 5;
            int kk = idx & 31;
            As[r][kk] = (_Float16)A[(size_t)(tileM + r) * K + (k0 + kk)];
        }
        __syncthreads();
        if (active) {
            // prefetch all 4 B fragments (8x global_load_b128 in one clause)
            const _Float16* wkt = Wp + (size_t)(k0 >> 5) * Nc * 32;
            v16h bfrag[4];
#pragma unroll
            for (int t = 0; t < 4; ++t) {
                const _Float16* base = wkt + (size_t)(tileN + t * 16 + lr) * 32 + lg * 16;
                v8h b0 = *(const v8h*)(base);
                v8h b1 = *(const v8h*)(base + 8);
#pragma unroll
                for (int i = 0; i < 8; ++i) { bfrag[t][i] = b0[i]; bfrag[t][8 + i] = b1[i]; }
            }
            // A fragment: contiguous k = 16*lg + i (matches packed B)
            v16h a;
#pragma unroll
            for (int i = 0; i < 16; ++i) {
                int kidx = lg * 16 + i;
                a[i] = As[lr][kidx];
            }
            // 4 WMMAs back-to-back
#pragma unroll
            for (int t = 0; t < 4; ++t) {
                acc[t] = __builtin_amdgcn_wmma_f32_16x16x32_f16(
                    false, a, false, bfrag[t], (short)0, acc[t], false, false);
            }
        }
        __syncthreads();
    }
    if (active) {
#pragma unroll
        for (int t = 0; t < 4; ++t) {
            int colBase = tileN + t * 16;
#pragma unroll
            for (int j = 0; j < 8; ++j) {
                int row = tileM + lg * 8 + j;  // C/D 16x16 f32 layout
                C[(size_t)row * Nc + colBase + lr] = acc[t][j];
            }
        }
    }
}

// ---------- per-node dots: a_s[n] = h[n,:]@avs, a_d[n] = h[n,:]@avd --------
__global__ void node_dots(const float* __restrict__ h, const float* __restrict__ avs,
                          const float* __restrict__ avd, float* __restrict__ as_,
                          float* __restrict__ ad_, int N, int C) {
    int n = blockIdx.x * (blockDim.x >> 5) + (threadIdx.x >> 5);
    int lane = threadIdx.x & 31;
    if (n >= N) return;
    float s1 = 0.f, s2 = 0.f;
    for (int c = lane * 4; c < C; c += 128) {
        float4 hv = *(const float4*)(h + (size_t)n * C + c);
        float4 vs = *(const float4*)(avs + c);
        float4 vd = *(const float4*)(avd + c);
        s1 += hv.x * vs.x + hv.y * vs.y + hv.z * vs.z + hv.w * vs.w;
        s2 += hv.x * vd.x + hv.y * vd.y + hv.z * vd.z + hv.w * vd.w;
    }
#pragma unroll
    for (int off = 16; off; off >>= 1) {
        s1 += __shfl_xor(s1, off, 32);
        s2 += __shfl_xor(s2, off, 32);
    }
    if (lane == 0) { as_[n] = s1; ad_[n] = s2; }
}

// ---------- edge_attr column sums (for self-loop fill_value='mean') --------
__global__ void zero_small(float* p, int n) {
    int i = blockIdx.x * blockDim.x + threadIdx.x;
    if (i < n) p[i] = 0.f;
}

__global__ void edge_attr_sum(const float* __restrict__ ea, float* __restrict__ sum8, int E) {
    __shared__ float red[8];
    float acc[8] = {0, 0, 0, 0, 0, 0, 0, 0};
    int stride = gridDim.x * blockDim.x;
    for (int e = blockIdx.x * blockDim.x + threadIdx.x; e < E; e += stride) {
        const float* ep = ea + (size_t)e * 8;
#pragma unroll
        for (int k = 0; k < 8; ++k) acc[k] += ep[k];
    }
    if (threadIdx.x < 8) red[threadIdx.x] = 0.f;
    __syncthreads();
#pragma unroll
    for (int k = 0; k < 8; ++k) atomicAdd(&red[k], acc[k]);
    __syncthreads();
    if (threadIdx.x < 8) atomicAdd(&sum8[threadIdx.x], red[threadIdx.x]);
}

// ---------- cvec[k] = We[k,:]@aev ; aeloop = (sum8/E)@cvec ------------------
__global__ void cvec_kernel(const float* __restrict__ We, const float* __restrict__ aev,
                            const float* __restrict__ sum8, float invE,
                            float* __restrict__ cvec, float* __restrict__ aeloop, int C) {
    __shared__ float cv[8];
    int w = threadIdx.x >> 5, lane = threadIdx.x & 31;
    float s = 0.f;
    for (int c = lane; c < C; c += 32) s += We[(size_t)w * C + c] * aev[c];
#pragma unroll
    for (int off = 16; off; off >>= 1) s += __shfl_xor(s, off, 32);
    if (lane == 0) cv[w] = s;
    __syncthreads();
    if (threadIdx.x == 0) {
        float t = 0.f;
#pragma unroll
        for (int k = 0; k < 8; ++k) { cvec[k] = cv[k]; t += sum8[k] * invE * cv[k]; }
        *aeloop = t;
    }
}

// ---------- layer init: out = bias (broadcast rows), amax=-inf, den=0 ------
__global__ void init_layer(float* __restrict__ out, const float* __restrict__ bias,
                           unsigned* __restrict__ amax, float* __restrict__ den,
                           int N, int C) {
    int i = blockIdx.x * blockDim.x + threadIdx.x;
    if (i < N * C) out[i] = bias[i % C];
    if (i < N) { amax[i] = ENC_NEGINF; den[i] = 0.f; }
}

// ---------- pass 1: segment max of alpha over dst --------------------------
__global__ void edge_pass_max(const int* __restrict__ src, const int* __restrict__ dst,
                              const float* __restrict__ ea, const float* __restrict__ cvec,
                              const float* __restrict__ aeloopP,
                              const float* __restrict__ as_, const float* __restrict__ ad_,
                              unsigned* __restrict__ amax, int E, int N) {
    int e = blockIdx.x * blockDim.x + threadIdx.x;
    if (e >= E + N) return;
    int s, d;
    float a = edge_alpha(e, E, src, dst, ea, cvec, *aeloopP, as_, ad_, &s, &d);
    atomicMax(&amax[d], fenc(a));
}

// ---------- pass 2: denom = segment sum of exp(alpha - amax) ---------------
__global__ void edge_pass_den(const int* __restrict__ src, const int* __restrict__ dst,
                              const float* __restrict__ ea, const float* __restrict__ cvec,
                              const float* __restrict__ aeloopP,
                              const float* __restrict__ as_, const float* __restrict__ ad_,
                              const unsigned* __restrict__ amax, float* __restrict__ den,
                              int E, int N) {
    int e = blockIdx.x * blockDim.x + threadIdx.x;
    if (e >= E + N) return;
    int s, d;
    float a = edge_alpha(e, E, src, dst, ea, cvec, *aeloopP, as_, ad_, &s, &d);
    atomicAdd(&den[d], __expf(a - fdec(amax[d])));
}

// ---------- pass 3: out[dst,:] += w * h[src,:]  (one wave per edge) --------
__global__ void edge_pass_agg(const int* __restrict__ src, const int* __restrict__ dst,
                              const float* __restrict__ ea, const float* __restrict__ cvec,
                              const float* __restrict__ aeloopP,
                              const float* __restrict__ as_, const float* __restrict__ ad_,
                              const unsigned* __restrict__ amax, const float* __restrict__ den,
                              const float* __restrict__ h, float* __restrict__ out,
                              int E, int N, int C) {
    int e = blockIdx.x * (blockDim.x >> 5) + (threadIdx.x >> 5);
    int lane = threadIdx.x & 31;
    if (e >= E + N) return;
    int s = 0, d = 0; float wt = 0.f;
    if (lane == 0) {
        float a = edge_alpha(e, E, src, dst, ea, cvec, *aeloopP, as_, ad_, &s, &d);
        wt = __expf(a - fdec(amax[d])) / den[d];
    }
    s  = __shfl(s, 0, 32);
    d  = __shfl(d, 0, 32);
    wt = __shfl(wt, 0, 32);
    const float* hp = h + (size_t)s * C;
    float* op = out + (size_t)d * C;
    for (int c = lane * 4; c < C; c += 128) {
        float4 hv = *(const float4*)(hp + c);   // global_load_b128 gather
        atomicAdd(op + c + 0, hv.x * wt);
        atomicAdd(op + c + 1, hv.y * wt);
        atomicAdd(op + c + 2, hv.z * wt);
        atomicAdd(op + c + 3, hv.w * wt);
    }
}

__global__ void relu_kernel(float* __restrict__ x, int n) {
    int i = blockIdx.x * blockDim.x + threadIdx.x;
    if (i < n) { float v = x[i]; x[i] = v > 0.f ? v : 0.f; }
}

extern "C" void kernel_launch(void* const* d_in, const int* in_sizes, int n_in,
                              void* d_out, int out_size, void* d_ws, size_t ws_size,
                              hipStream_t stream) {
    (void)n_in; (void)out_size; (void)ws_size;
    const float* x   = (const float*)d_in[0];
    const int*   ei  = (const int*)d_in[1];
    const float* ea  = (const float*)d_in[2];
    const float* W1  = (const float*)d_in[3];
    const float* We1 = (const float*)d_in[4];
    const float* as1 = (const float*)d_in[5];
    const float* ad1 = (const float*)d_in[6];
    const float* ae1 = (const float*)d_in[7];
    const float* b1  = (const float*)d_in[8];
    const float* W2  = (const float*)d_in[9];
    const float* We2 = (const float*)d_in[10];
    const float* as2 = (const float*)d_in[11];
    const float* ad2 = (const float*)d_in[12];
    const float* ae2 = (const float*)d_in[13];
    const float* b2  = (const float*)d_in[14];

    const int E    = in_sizes[1] / 2;
    const int C_h  = in_sizes[5];              // 256
    const int C_in = in_sizes[3] / C_h;        // 128
    const int C_o  = in_sizes[11];             // 128
    const int N    = in_sizes[0] / C_in;       // 50000
    const int ET   = E + N;

    const int* src = ei;
    const int* dst = ei + E;

    // ---- workspace layout ----
    float*    H    = (float*)d_ws;             // N * C_h (reused for h2)
    float*    O1   = H + (size_t)N * C_h;      // N * C_h
    float*    AS   = O1 + (size_t)N * C_h;     // N
    float*    AD   = AS + N;                   // N
    unsigned* AMAX = (unsigned*)(AD + N);      // N
    float*    DEN  = (float*)(AMAX + N);       // N
    float*    SUM8 = DEN + N;                  // 8
    float*    CVEC = SUM8 + 8;                 // 8
    float*    AEL  = CVEC + 8;                 // 1
    _Float16* WP   = (_Float16*)(AEL + 1);     // max(K*Nc) halves (<= 64KB)

    float* out2 = (float*)d_out;
    const float invE = 1.0f / (float)E;

    // edge_attr column sums (shared by both layers)
    zero_small<<<1, 32, 0, stream>>>(SUM8, 8);
    edge_attr_sum<<<64, 256, 0, stream>>>(ea, SUM8, E);

    // =================== layer 1 ===================
    cvec_kernel<<<1, 256, 0, stream>>>(We1, ae1, SUM8, invE, CVEC, AEL, C_h);
    pack_w_f16<<<(C_in * C_h + 255) / 256, 256, 0, stream>>>(W1, WP, C_in, C_h);
    gemm_wmma_f16<<<(N + 15) / 16, 128, 0, stream>>>(x, WP, H, N, C_in, C_h);
    node_dots<<<(N + 7) / 8, 256, 0, stream>>>(H, as1, ad1, AS, AD, N, C_h);
    init_layer<<<((size_t)N * C_h + 255) / 256, 256, 0, stream>>>(O1, b1, AMAX, DEN, N, C_h);
    edge_pass_max<<<(ET + 255) / 256, 256, 0, stream>>>(src, dst, ea, CVEC, AEL, AS, AD, AMAX, E, N);
    edge_pass_den<<<(ET + 255) / 256, 256, 0, stream>>>(src, dst, ea, CVEC, AEL, AS, AD, AMAX, DEN, E, N);
    edge_pass_agg<<<(ET + 7) / 8, 256, 0, stream>>>(src, dst, ea, CVEC, AEL, AS, AD, AMAX, DEN, H, O1, E, N, C_h);
    relu_kernel<<<((size_t)N * C_h + 255) / 256, 256, 0, stream>>>(O1, N * C_h);

    // =================== layer 2 ===================
    cvec_kernel<<<1, 256, 0, stream>>>(We2, ae2, SUM8, invE, CVEC, AEL, C_o);
    pack_w_f16<<<(C_h * C_o + 255) / 256, 256, 0, stream>>>(W2, WP, C_h, C_o);
    gemm_wmma_f16<<<(N + 15) / 16, 128, 0, stream>>>(O1, WP, H, N, C_h, C_o);
    node_dots<<<(N + 7) / 8, 256, 0, stream>>>(H, as2, ad2, AS, AD, N, C_o);
    init_layer<<<((size_t)N * C_o + 255) / 256, 256, 0, stream>>>(out2, b2, AMAX, DEN, N, C_o);
    edge_pass_max<<<(ET + 255) / 256, 256, 0, stream>>>(src, dst, ea, CVEC, AEL, AS, AD, AMAX, E, N);
    edge_pass_den<<<(ET + 255) / 256, 256, 0, stream>>>(src, dst, ea, CVEC, AEL, AS, AD, AMAX, DEN, E, N);
    edge_pass_agg<<<(ET + 7) / 8, 256, 0, stream>>>(src, dst, ea, CVEC, AEL, AS, AD, AMAX, DEN, H, out2, E, N, C_o);
}
